// DotProductAttention_20186346292012
// MI455X (gfx1250) — compile-verified, matrix operand order checked by reading, not run on hardware
//
#include <hip/hip_runtime.h>
#include <stdint.h>

// ---------------------------------------------------------------------------
// CDNA5 (gfx1250) attention pipeline, bf16 WMMA, fp32 accumulate.
// Phase 0: convert q, x, Wq, Wk, Wv fp32 -> bf16 once (bandwidth-bound).
// Phase 1: qp = q@Wq^T+bq ; k = x@Wk^T+bk ; vT = (x@Wv^T+bv)^T  (bf16 GEMMs)
// Phase 2 (per batch, score matrix stays in 192MB L2):
//          S = qp@k^T * 1/32 (fp32) ; P = softmax(S) (bf16) ; out = P@vT^T.
// All GEMMs NT-form: operands row-major [rows, K], K contiguous, so every
// WMMA fragment is two 16-byte per-lane GLOBAL loads (address_space(1) forced
// via POD ext_vector types so the compiler emits global_load_b128, not
// flat_load_b128 -> no DScnt coupling, LDS pipe stays free).
// Wave tile 64x64 -> 16 v_wmma : 16 loads per k-step.
// ---------------------------------------------------------------------------

typedef __attribute__((ext_vector_type(16))) __bf16       v16bf;
typedef __attribute__((ext_vector_type(8)))  float        v8f;
typedef __attribute__((ext_vector_type(4)))  unsigned int u32x4;
typedef __attribute__((ext_vector_type(4)))  float        f32x4;

union FragU {
    uint32_t u[8];
    u32x4    q[2];
    v16bf    v;
};

// --- address-space(1) access helpers: force global_load/global_store -------
__device__ __forceinline__ u32x4 gload_b128(const void* p) {
    typedef __attribute__((address_space(1))) const u32x4* gp_t;
    return *(gp_t)(unsigned long long)p;
}
__device__ __forceinline__ f32x4 gload_f128(const void* p) {
    typedef __attribute__((address_space(1))) const f32x4* gp_t;
    return *(gp_t)(unsigned long long)p;
}
__device__ __forceinline__ float gload_f32(const void* p) {
    typedef __attribute__((address_space(1))) const float* gp_t;
    return *(gp_t)(unsigned long long)p;
}
__device__ __forceinline__ void gstore_b128(void* p, u32x4 v) {
    typedef __attribute__((address_space(1))) u32x4* gp_t;
    *(gp_t)(unsigned long long)p = v;
}
__device__ __forceinline__ void gstore_f32(void* p, float v) {
    typedef __attribute__((address_space(1))) float* gp_t;
    *(gp_t)(unsigned long long)p = v;
}
__device__ __forceinline__ void gstore_u16(void* p, unsigned short v) {
    typedef __attribute__((address_space(1))) unsigned short* gp_t;
    *(gp_t)(unsigned long long)p = v;
}

// Round-to-nearest fp32->bf16, two at a time, packed with one v_perm_b32.
__device__ __forceinline__ uint32_t pack2_bf16(float x, float y) {
    uint32_t ux = __float_as_uint(x) + 0x8000u;
    uint32_t uy = __float_as_uint(y) + 0x8000u;
    return __builtin_amdgcn_perm(uy, ux, 0x07060302u);  // {uy[31:16], ux[31:16]}
}

__device__ __forceinline__ unsigned short f32_to_bf16(float a) {
    uint32_t ua = __float_as_uint(a) + 0x8000u;
    return (unsigned short)(ua >> 16);
}

// ---------------------------------------------------------------------------
// fp32 -> bf16 bulk conversion: 8 elements per thread, 128b in / 128b out.
// ---------------------------------------------------------------------------
__global__ __launch_bounds__(256)
void cvt_f32_to_bf16(const float* __restrict__ in, unsigned short* __restrict__ out, int n)
{
    const int i = (blockIdx.x * 256 + threadIdx.x) * 8;
    if (i >= n) return;
    f32x4 a = gload_f128(in + i);
    f32x4 b = gload_f128(in + i + 4);
    u32x4 r;
    r.x = pack2_bf16(a.x, a.y);
    r.y = pack2_bf16(a.z, a.w);
    r.z = pack2_bf16(b.x, b.y);
    r.w = pack2_bf16(b.z, b.w);
    gstore_b128(out + i, r);
}

// ---------------------------------------------------------------------------
// Shared bf16 NT GEMM core: wave tile 64x64 (4x4 WMMA frags), block 8 waves
// arranged 2x4 -> block tile 128x256. acc must be zero-initialized by caller.
// ---------------------------------------------------------------------------
struct WaveCoord {
    int mBase, nBase, hb, lrow;
};

__device__ __forceinline__ WaveCoord wave_coord() {
    WaveCoord c;
    const int lane  = threadIdx.x & 31;
    const int wave  = threadIdx.x >> 5;
    c.hb    = lane >> 4;
    c.lrow  = lane & 15;
    const int waveM = wave & 1;
    const int waveN = wave >> 1;
    c.mBase = blockIdx.x * 128 + waveM * 64;
    c.nBase = blockIdx.y * 256 + waveN * 64;
    return c;
}

__device__ __forceinline__ void gemm_core_bf16(
    const unsigned short* __restrict__ A, const unsigned short* __restrict__ Bm,
    int K, const WaveCoord& c, v8f acc[4][4])
{
    const unsigned short* aPtr[4];
    const unsigned short* bPtr[4];
    #pragma unroll
    for (int i = 0; i < 4; ++i)
        aPtr[i] = A + (size_t)(c.mBase + i * 16 + c.lrow) * K + c.hb * 8;
    #pragma unroll
    for (int j = 0; j < 4; ++j)
        bPtr[j] = Bm + (size_t)(c.nBase + j * 16 + c.lrow) * K + c.hb * 8;

    for (int k0 = 0; k0 < K; k0 += 32) {
        FragU a[4], b[4];
        #pragma unroll
        for (int i = 0; i < 4; ++i) {
            a[i].q[0] = gload_b128(aPtr[i]);
            a[i].q[1] = gload_b128(aPtr[i] + 16);
            aPtr[i] += 32;
        }
        #pragma unroll
        for (int j = 0; j < 4; ++j) {
            b[j].q[0] = gload_b128(bPtr[j]);
            b[j].q[1] = gload_b128(bPtr[j] + 16);
            bPtr[j] += 32;
        }
        #pragma unroll
        for (int i = 0; i < 4; ++i)
            #pragma unroll
            for (int j = 0; j < 4; ++j)
                acc[i][j] = __builtin_amdgcn_wmma_f32_16x16x32_bf16(
                    false, a[i].v, false, b[j].v, (short)0, acc[i][j], false, false);
    }
}

// ---------------------------------------------------------------------------
// Projection GEMM: C = A(bf16)[M,K] @ W(bf16)[N,K]^T + bias[N], bf16 out.
// transposed=1 stores out[b, n, s] (per-batch transpose, for V).
// ---------------------------------------------------------------------------
__global__ __launch_bounds__(256)
void gemm_bf16_bf16out(const unsigned short* __restrict__ A,
                       const unsigned short* __restrict__ W,
                       const float* __restrict__ bias,
                       unsigned short* __restrict__ out,
                       int N, int K, int Srows, int transposed)
{
    const WaveCoord c = wave_coord();
    v8f acc[4][4];
    #pragma unroll
    for (int i = 0; i < 4; ++i)
        #pragma unroll
        for (int j = 0; j < 4; ++j)
            #pragma unroll
            for (int t = 0; t < 8; ++t) acc[i][j][t] = 0.0f;

    gemm_core_bf16(A, W, K, c, acc);

    #pragma unroll
    for (int j = 0; j < 4; ++j) {
        const int n  = c.nBase + j * 16 + c.lrow;
        const float bn = gload_f32(bias + n);
        #pragma unroll
        for (int i = 0; i < 4; ++i) {
            #pragma unroll
            for (int r = 0; r < 8; ++r) {
                const int m = c.mBase + i * 16 + c.hb * 8 + r;
                const unsigned short h = f32_to_bf16(acc[i][j][r] + bn);
                if (!transposed) {
                    gstore_u16(out + (size_t)m * N + n, h);
                } else {
                    const int bb = m / Srows, s = m % Srows;
                    gstore_u16(out + (size_t)bb * N * Srows + (size_t)n * Srows + s, h);
                }
            }
        }
    }
}

// ---------------------------------------------------------------------------
// bf16 NT GEMM, fp32 out with scale (QK^T and P@V).
// ---------------------------------------------------------------------------
__global__ __launch_bounds__(256)
void gemm_bf16_f32out(const unsigned short* __restrict__ A,
                      const unsigned short* __restrict__ Bm,
                      float* __restrict__ C,
                      int K, int ldc, float scale)
{
    const WaveCoord c = wave_coord();
    v8f acc[4][4];
    #pragma unroll
    for (int i = 0; i < 4; ++i)
        #pragma unroll
        for (int j = 0; j < 4; ++j)
            #pragma unroll
            for (int t = 0; t < 8; ++t) acc[i][j][t] = 0.0f;

    gemm_core_bf16(A, Bm, K, c, acc);

    #pragma unroll
    for (int j = 0; j < 4; ++j) {
        const int n = c.nBase + j * 16 + c.lrow;
        #pragma unroll
        for (int i = 0; i < 4; ++i) {
            #pragma unroll
            for (int r = 0; r < 8; ++r) {
                const int m = c.mBase + i * 16 + c.hb * 8 + r;
                gstore_f32(C + (size_t)m * ldc + n, acc[i][j][r] * scale);
            }
        }
    }
}

// ---------------------------------------------------------------------------
// Row softmax: fp32 scores row -> bf16 probabilities. One 256-thread block
// per row, 8 strided elements per thread (ncols = 2048).
// ---------------------------------------------------------------------------
__global__ __launch_bounds__(256)
void softmax_bf16(const float* __restrict__ S, unsigned short* __restrict__ P, int ncols)
{
    __shared__ float red[256];
    const int row = blockIdx.x;
    const int tid = threadIdx.x;
    const float* srow = S + (size_t)row * ncols;
    const int per = ncols / 256;   // 8

    float vals[8];
    float vmax = -3.0e38f;
    for (int t = 0; t < per; ++t) {
        vals[t] = gload_f32(srow + tid + t * 256);
        vmax = fmaxf(vmax, vals[t]);
    }
    red[tid] = vmax; __syncthreads();
    for (int s = 128; s > 0; s >>= 1) {
        if (tid < s) red[tid] = fmaxf(red[tid], red[tid + s]);
        __syncthreads();
    }
    vmax = red[0]; __syncthreads();

    float vsum = 0.0f;
    for (int t = 0; t < per; ++t) {
        vals[t] = __expf(vals[t] - vmax);
        vsum += vals[t];
    }
    red[tid] = vsum; __syncthreads();
    for (int s = 128; s > 0; s >>= 1) {
        if (tid < s) red[tid] += red[tid + s];
        __syncthreads();
    }
    const float inv = 1.0f / red[0];

    unsigned short* prow = P + (size_t)row * ncols;
    for (int t = 0; t < per; ++t)
        gstore_u16(prow + tid + t * 256, f32_to_bf16(vals[t] * inv));
}

// ---------------------------------------------------------------------------
extern "C" void kernel_launch(void* const* d_in, const int* in_sizes, int n_in,
                              void* d_out, int out_size, void* d_ws, size_t ws_size,
                              hipStream_t stream)
{
    (void)in_sizes; (void)n_in; (void)out_size; (void)ws_size;
    const int Bn = 4, SQ = 2048, SKV = 2048, D = 1024, H = 1024;
    const size_t MB = 1024 * 1024;

    const float* q  = (const float*)d_in[0];
    const float* x  = (const float*)d_in[1];
    const float* Wq = (const float*)d_in[2];
    const float* bq = (const float*)d_in[3];
    const float* Wk = (const float*)d_in[4];
    const float* bk = (const float*)d_in[5];
    const float* Wv = (const float*)d_in[6];
    const float* bv = (const float*)d_in[7];
    float* out = (float*)d_out;

    char* ws = (char*)d_ws;
    unsigned short* qbf = (unsigned short*)(ws);             // [B*SQ, D]   bf16 16MB
    unsigned short* xbf = (unsigned short*)(ws + 16 * MB);   // [B*SKV, D]  bf16 16MB
    unsigned short* wqb = (unsigned short*)(ws + 32 * MB);   // [H, D]      bf16  2MB
    unsigned short* wkb = (unsigned short*)(ws + 34 * MB);   // [H, D]      bf16  2MB
    unsigned short* wvb = (unsigned short*)(ws + 36 * MB);   // [H, D]      bf16  2MB
    unsigned short* qp  = (unsigned short*)(ws + 40 * MB);   // [B*SQ, H]   bf16 16MB
    unsigned short* kp  = (unsigned short*)(ws + 56 * MB);   // [B*SKV, H]  bf16 16MB
    unsigned short* vT  = (unsigned short*)(ws + 72 * MB);   // [B, H, SKV] bf16 16MB
    float*          sc  = (float*)         (ws + 88 * MB);   // [SQ, SKV]   f32  16MB
    unsigned short* at  = (unsigned short*)(ws + 104 * MB);  // [SQ, SKV]   bf16  8MB

    dim3 blk(256);

    // Phase 0: bulk fp32 -> bf16 conversion (bandwidth-bound, ~76MB total).
    const int nAct = Bn * SQ * D;       // 8M elements
    const int nW   = H * D;             // 1M elements
    hipLaunchKernelGGL(cvt_f32_to_bf16, dim3(nAct / (8 * 256)), blk, 0, stream, q,  qbf, nAct);
    hipLaunchKernelGGL(cvt_f32_to_bf16, dim3(nAct / (8 * 256)), blk, 0, stream, x,  xbf, nAct);
    hipLaunchKernelGGL(cvt_f32_to_bf16, dim3(nW   / (8 * 256)), blk, 0, stream, Wq, wqb, nW);
    hipLaunchKernelGGL(cvt_f32_to_bf16, dim3(nW   / (8 * 256)), blk, 0, stream, Wk, wkb, nW);
    hipLaunchKernelGGL(cvt_f32_to_bf16, dim3(nW   / (8 * 256)), blk, 0, stream, Wv, wvb, nW);

    // Phase 1: projections, batches merged along M = B*S. Block tile 128x256.
    dim3 gproj((Bn * SQ) / 128, H / 256);
    hipLaunchKernelGGL(gemm_bf16_bf16out, gproj, blk, 0, stream, qbf, wqb, bq, qp, H, D, SQ,  0);
    hipLaunchKernelGGL(gemm_bf16_bf16out, gproj, blk, 0, stream, xbf, wkb, bk, kp, H, D, SKV, 0);
    hipLaunchKernelGGL(gemm_bf16_bf16out, gproj, blk, 0, stream, xbf, wvb, bv, vT, H, D, SKV, 1);

    // Phase 2: per-batch attention; score matrix (16MB) stays L2-resident.
    const float scale = 0.03125f;   // 1/sqrt(1024)
    for (int b = 0; b < Bn; ++b) {
        const unsigned short* qb = qp + (size_t)b * SQ  * H;
        const unsigned short* kb = kp + (size_t)b * SKV * H;
        const unsigned short* vb = vT + (size_t)b * H   * SKV;
        float* ob = out + (size_t)b * SQ * H;

        dim3 gs(SQ / 128, SKV / 256);
        hipLaunchKernelGGL(gemm_bf16_f32out, gs, blk, 0, stream, qb, kb, sc, H, SKV, scale);
        hipLaunchKernelGGL(softmax_bf16, dim3(SQ), blk, 0, stream, sc, at, SKV);
        dim3 gc(SQ / 128, H / 256);
        hipLaunchKernelGGL(gemm_bf16_f32out, gc, blk, 0, stream, at, vb, ob, SQ, H, 1.0f);
    }
}
